// ChatDecoder_87393994539549
// MI455X (gfx1250) — matrix operand order repaired and checked by reading, not compiled, not run on hardware
//
#include <hip/hip_runtime.h>
#include <hip/hip_bf16.h>

// ---------------------------------------------------------------------------
// ChatDecoder (GRU + dot attention + BN + vocab softmax) for MI455X / gfx1250
// GEMMs via v_wmma_f32_16x16x32_bf16 (wave32 WMMA), tiled through LDS.
// Weights pre-converted to bf16 (W1/W2 transposed to [N,K]) once per launch;
// hot-loop tile staging is 16B global loads -> 16B LDS stores. GEMM is
// templated on (A-staging, B-staging, gather, epilogue) so every
// instantiation has a branch-free inner loop.
// ---------------------------------------------------------------------------

typedef __bf16 bf16;
typedef __attribute__((ext_vector_type(16))) __bf16 v16bf;
typedef __attribute__((ext_vector_type(8)))  __bf16 v8bf;
typedef __attribute__((ext_vector_type(8)))  float  v8f;

#define BB   64      // batch
#define TT   32      // time steps
#define HH   1024    // hidden
#define EE   512     // embedding
#define VV   32000   // vocab
#define SS   64      // encoder seq len

#define BM   64
#define BN   64
#define BKK  32
#define PADK 8       // LDS row pad: stride 40 bf16 = 80B (multiple of 16B)

// ---------------------------------------------------------------------------
// Tiled WMMA GEMM:  C[M,N] = act( A[M,K] @ B[K,N] + bias[N] )
//   ABF    : 1 -> A staged from bf16 copy Abf (same lda)
//   BBF    : 1 -> B staged from bf16 copy stored [N,K] with ld = K
//   GATHER : A row m is row rowIdx[m] (embedding gather)
//   EPI    : 0 = bias ; 1 = bias + ELU(alpha=1)
// M,N multiples of 64; K multiple of 32 (true for all call sites).
// ---------------------------------------------------------------------------
template <int ABF, int BBF, int GATHER, int EPI>
__global__ __launch_bounds__(128)
void gemm_wmma_kernel(const float* __restrict__ A, int lda,
                      const int* __restrict__ rowIdx,
                      const bf16* __restrict__ Abf,
                      const float* __restrict__ Bmat, int ldb, int bTrans,
                      const bf16* __restrict__ BbfT,
                      const float* __restrict__ bias,
                      float* __restrict__ C, int ldc,
                      int M, int N, int K)
{
    __shared__ bf16 As[BM][BKK + PADK];   // [m][k]
    __shared__ bf16 Bs[BN][BKK + PADK];   // [n][k] (K-contiguous per column)

    const int tid  = threadIdx.x;
    const int lane = tid & 31;
    const int wave = tid >> 5;
    const int wr   = wave >> 1;           // wave row (0..1)
    const int wc   = wave & 1;            // wave col (0..1)
    const int half = lane >> 4;           // lane half (A/B K-split, C M-split)
    const int l16  = lane & 15;

    const int rowBase = blockIdx.x * BM;
    const int colBase = blockIdx.y * BN;

    v8f acc[2][2] = {};

    for (int k0 = 0; k0 < K; k0 += BKK) {
        // -------- A tile 64x32 --------
        if (ABF) {
            #pragma unroll
            for (int i = 0; i < 2; ++i) {
                int idx = (tid + i * 128) * 8;          // 8 contiguous elems
                int m = idx >> 5, kk = idx & 31;
                long grow = rowBase + m;
                long ar = GATHER ? (long)rowIdx[grow] : grow;
                v8bf v = *(const v8bf*)(Abf + ar * lda + (k0 + kk));
                *(v8bf*)&As[m][kk] = v;
            }
        } else {
            #pragma unroll
            for (int i = 0; i < 16; ++i) {
                int idx = tid + i * 128;
                int m = idx >> 5, kk = idx & 31;
                long grow = rowBase + m;
                long ar = GATHER ? (long)rowIdx[grow] : grow;
                As[m][kk] = (bf16)A[ar * lda + (k0 + kk)];
            }
        }
        // -------- B tile 32x64, stored as Bs[n][k] --------
        if (BBF) {
            #pragma unroll
            for (int i = 0; i < 2; ++i) {
                int idx = (tid + i * 128) * 8;          // 8 contiguous along k
                int n = idx >> 5, kk = idx & 31;
                v8bf v = *(const v8bf*)(BbfT + (long)(colBase + n) * K + (k0 + kk));
                *(v8bf*)&Bs[n][kk] = v;
            }
            if (k0 + BKK < K)   // prefetch next K-panel (global_prefetch_b8)
                __builtin_prefetch(BbfT + (long)(colBase + (tid >> 1)) * K + (k0 + BKK), 0, 0);
        } else {
            #pragma unroll
            for (int i = 0; i < 16; ++i) {
                int idx = tid + i * 128;
                int kk, n;
                if (bTrans) { n = idx >> 5; kk = idx & 31; }
                else        { kk = idx >> 6; n = idx & 63; }
                long off = bTrans ? ((long)(colBase + n) * ldb + (k0 + kk))
                                  : ((long)(k0 + kk) * ldb + (colBase + n));
                Bs[n][kk] = (bf16)Bmat[off];
            }
        }
        __syncthreads();

        // -------- pack fragments per ISA 7.12.2 wave32 layouts --------
        v16bf afrag[2], bfrag[2];
        #pragma unroll
        for (int i = 0; i < 2; ++i) {
            int m = wr * 32 + i * 16 + l16;
            #pragma unroll
            for (int e = 0; e < 8; ++e) afrag[i][e]     = As[m][half * 8 + e];
            #pragma unroll
            for (int e = 0; e < 8; ++e) afrag[i][8 + e] = As[m][16 + half * 8 + e];
        }
        #pragma unroll
        for (int j = 0; j < 2; ++j) {
            int n = wc * 32 + j * 16 + l16;
            #pragma unroll
            for (int e = 0; e < 16; ++e) bfrag[j][e] = Bs[n][(e >> 3) * 16 + half * 8 + (e & 7)];
        }

        #pragma unroll
        for (int i = 0; i < 2; ++i)
            #pragma unroll
            for (int j = 0; j < 2; ++j)
                acc[i][j] = __builtin_amdgcn_wmma_f32_16x16x32_bf16(
                    false, afrag[i], false, bfrag[j],
                    (short)0, acc[i][j], false, false);
        __syncthreads();
    }

    // -------- epilogue + store (C layout: M = half*8 + r, N = l16) --------
    #pragma unroll
    for (int i = 0; i < 2; ++i)
        #pragma unroll
        for (int j = 0; j < 2; ++j)
            #pragma unroll
            for (int r = 0; r < 8; ++r) {
                int grow = rowBase + wr * 32 + i * 16 + half * 8 + r;
                int gcol = colBase + wc * 32 + j * 16 + l16;
                float c = acc[i][j][r] + bias[gcol];
                if (EPI == 1) c = c > 0.f ? c : expm1f(c);   // ELU
                C[(long)grow * ldc + gcol] = c;
            }
}

// ---------------------------------------------------------------------------
// f32 -> bf16 straight conversion (n multiple of 8)
// ---------------------------------------------------------------------------
__global__ void cvt_bf16_kernel(const float* __restrict__ src,
                                bf16* __restrict__ dst, long n)
{
    long i = ((long)blockIdx.x * blockDim.x + threadIdx.x) * 8;
    if (i >= n) return;
    float4 a = *(const float4*)(src + i);
    float4 b = *(const float4*)(src + i + 4);
    v8bf v;
    v[0] = (bf16)a.x; v[1] = (bf16)a.y; v[2] = (bf16)a.z; v[3] = (bf16)a.w;
    v[4] = (bf16)b.x; v[5] = (bf16)b.y; v[6] = (bf16)b.z; v[7] = (bf16)b.w;
    *(v8bf*)(dst + i) = v;
}

// ---------------------------------------------------------------------------
// f32 [K,N] -> bf16 [N,K] transpose-convert via 32x32 LDS tiles
// ---------------------------------------------------------------------------
__global__ __launch_bounds__(256)
void cvtT_bf16_kernel(const float* __restrict__ src,
                      bf16* __restrict__ dst, int K, int N)
{
    __shared__ float tile[32][33];
    int kb = blockIdx.x * 32, nb = blockIdx.y * 32;
    int tx = threadIdx.x & 31, ty = threadIdx.x >> 5;    // 32 x 8
    #pragma unroll
    for (int r = 0; r < 32; r += 8)
        tile[ty + r][tx] = src[(long)(kb + ty + r) * N + (nb + tx)];
    __syncthreads();
    #pragma unroll
    for (int r = 0; r < 32; r += 8)
        dst[(long)(nb + ty + r) * K + (kb + tx)] = (bf16)tile[tx][ty + r];
}

// ---------------------------------------------------------------------------
__global__ void build_tokens_kernel(const int* __restrict__ label,
                                    int* __restrict__ tokens)
{
    int i = blockIdx.x * blockDim.x + threadIdx.x;
    if (i >= TT * BB) return;
    int t = i / BB, b = i % BB;
    tokens[i] = (t == 0) ? 1 : label[b * TT + (t - 1)];
}

// ---------------------------------------------------------------------------
__global__ void gru_combine_kernel(const float* __restrict__ gx,  // [B,3H]
                                   const float* __restrict__ gh,  // [B,3H]
                                   float* __restrict__ h)         // [B,H]
{
    int i = blockIdx.x * blockDim.x + threadIdx.x;   // 0..B*H-1
    int b = i >> 10, c = i & (HH - 1);
    const float* gxb = gx + b * 3 * HH;
    const float* ghb = gh + b * 3 * HH;
    float xr = gxb[c], xz = gxb[HH + c], xn = gxb[2 * HH + c];
    float hr = ghb[c], hz = ghb[HH + c], hn = ghb[2 * HH + c];
    float r = 1.f / (1.f + expf(-(xr + hr)));
    float z = 1.f / (1.f + expf(-(xz + hz)));
    float n = tanhf(xn + r * hn);
    h[i] = (1.f - z) * n + z * h[i];
}

// ---------------------------------------------------------------------------
__global__ __launch_bounds__(256)
void attention_kernel(const float* __restrict__ h,    // [B,H]
                      const float* __restrict__ enc,  // [B,S,H]
                      float* __restrict__ feat)       // [B,2H]
{
    __shared__ float hrow[HH];
    __shared__ float sc[SS];
    __shared__ float wsm[SS];
    int b = blockIdx.x, tid = threadIdx.x;
    int lane = tid & 31, wave = tid >> 5;
    const float* hb = h + b * HH;
    const float* eb = enc + (long)b * SS * HH;

    for (int k = tid; k < HH; k += 256) hrow[k] = hb[k];
    __syncthreads();

    for (int q = 0; q < 8; ++q) {                 // 8 waves x 8 scores
        int s = wave * 8 + q;
        const float* er = eb + s * HH;
        float p = 0.f;
        for (int k = lane; k < HH; k += 32) p += hrow[k] * er[k];
        for (int off = 16; off > 0; off >>= 1) p += __shfl_xor(p, off, 32);
        if (lane == 0) sc[s] = p;
    }
    __syncthreads();
    if (tid == 0) {
        float mx = sc[0];
        for (int s = 1; s < SS; ++s) mx = fmaxf(mx, sc[s]);
        float sum = 0.f;
        for (int s = 0; s < SS; ++s) { float e = expf(sc[s] - mx); wsm[s] = e; sum += e; }
        float inv = 1.f / sum;
        for (int s = 0; s < SS; ++s) wsm[s] *= inv;
    }
    __syncthreads();

    float* fb = feat + b * 2 * HH;
    for (int k = tid; k < HH; k += 256) fb[k] = hrow[k];
    for (int c = tid; c < HH; c += 256) {
        float a = 0.f;
        for (int s = 0; s < SS; ++s) a += wsm[s] * eb[s * HH + c];
        fb[HH + c] = a;
    }
}

// ---------------------------------------------------------------------------
__global__ void bn_kernel(const float* __restrict__ x, float* __restrict__ y,
                          const float* __restrict__ g, const float* __restrict__ beta,
                          int C)
{
    int c = blockIdx.x * blockDim.x + threadIdx.x;
    if (c >= C) return;
    float s = 0.f, s2 = 0.f;
    for (int r = 0; r < BB; ++r) { float v = x[r * C + c]; s += v; s2 += v * v; }
    float mu  = s  * (1.f / BB);
    float var = s2 * (1.f / BB) - mu * mu;
    float k   = rsqrtf(var + 1e-5f) * g[c];
    float bb  = beta[c];
    for (int r = 0; r < BB; ++r) y[r * C + c] = (x[r * C + c] - mu) * k + bb;
}

// ---------------------------------------------------------------------------
__global__ __launch_bounds__(256)
void softmax_rows_kernel(const float* __restrict__ logits,  // [B,V]
                         float* __restrict__ out, int t)    // out [B,T,V]
{
    __shared__ float red[256];
    int b = blockIdx.x, tid = threadIdx.x;
    const float* lb = logits + (long)b * VV;

    float mx = -3.4e38f;
    for (int v = tid; v < VV; v += 256) mx = fmaxf(mx, lb[v]);
    red[tid] = mx; __syncthreads();
    for (int o = 128; o > 0; o >>= 1) {
        if (tid < o) red[tid] = fmaxf(red[tid], red[tid + o]);
        __syncthreads();
    }
    mx = red[0]; __syncthreads();

    float sum = 0.f;
    for (int v = tid; v < VV; v += 256) sum += expf(lb[v] - mx);
    red[tid] = sum; __syncthreads();
    for (int o = 128; o > 0; o >>= 1) {
        if (tid < o) red[tid] += red[tid + o];
        __syncthreads();
    }
    float inv = 1.f / red[0];

    float* ob = out + ((long)b * TT + t) * VV;
    for (int v = tid; v < VV; v += 256) ob[v] = expf(lb[v] - mx) * inv;
}

// ---------------------------------------------------------------------------
extern "C" void kernel_launch(void* const* d_in, const int* in_sizes, int n_in,
                              void* d_out, int out_size, void* d_ws, size_t ws_size,
                              hipStream_t stream)
{
    (void)in_sizes; (void)n_in; (void)out_size;

    const float* enc   = (const float*)d_in[0];
    const int*   label = (const int*)  d_in[1];
    const float* h0    = (const float*)d_in[2];
    const float* emb   = (const float*)d_in[4];   // [V,E]
    const float* w_ih  = (const float*)d_in[5];   // [3H,E] (= [N,K])
    const float* w_hh  = (const float*)d_in[6];   // [3H,H] (= [N,K])
    const float* b_ih  = (const float*)d_in[7];
    const float* b_hh  = (const float*)d_in[8];
    const float* bn1_g = (const float*)d_in[9];
    const float* bn1_b = (const float*)d_in[10];
    const float* W1    = (const float*)d_in[11];  // [2048,1024] (= [K,N])
    const float* b1    = (const float*)d_in[12];
    const float* bn2_g = (const float*)d_in[13];
    const float* bn2_b = (const float*)d_in[14];
    const float* W2    = (const float*)d_in[15];  // [1024,32000] (= [K,N])
    const float* b2    = (const float*)d_in[16];

    float* out = (float*)d_out;                   // [B,T,V] then [1,B,H]

    // ---- workspace layout ------------------------------------------------
    float* ws     = (float*)d_ws;
    int*   tokens = (int*)ws;                               // 2048 ints
    float* gx     = ws + 2048;                              // T*B*3H
    float* h      = gx + (long)TT * BB * 3 * HH;            // B*H
    float* gh     = h + BB * HH;                            // B*3H
    float* feat   = gh + BB * 3 * HH;                       // B*2H
    float* featn  = feat + BB * 2 * HH;                     // B*2H
    float* y      = featn + BB * 2 * HH;                    // B*H
    float* yn     = y + BB * HH;                            // B*H
    float* logits = yn + BB * HH;                           // B*V

    bf16* embBf = (bf16*)(logits + (long)BB * VV);          // [V,E]
    bf16* wihBf = embBf + (long)VV * EE;                    // [3H,E]
    bf16* whhBf = wihBf + (long)3 * HH * EE;                // [3H,H]
    bf16* W1T   = whhBf + (long)3 * HH * HH;                // [1024,2048]
    bf16* W2T   = W1T + (long)2 * HH * HH;                  // [32000,1024]
    size_t need = (size_t)((char*)(W2T + (long)VV * HH) - (char*)d_ws);
    const bool useBf = (ws_size >= need);

    hipMemcpyAsync(h, h0, (size_t)BB * HH * sizeof(float),
                   hipMemcpyDeviceToDevice, stream);
    build_tokens_kernel<<<(TT * BB + 255) / 256, 256, 0, stream>>>(label, tokens);

    if (useBf) {
        long nEmb = (long)VV * EE, nIh = (long)3 * HH * EE, nHh = (long)3 * HH * HH;
        cvt_bf16_kernel<<<(int)(nEmb / 8 / 256), 256, 0, stream>>>(emb,  embBf, nEmb);
        cvt_bf16_kernel<<<(int)(nIh  / 8 / 256), 256, 0, stream>>>(w_ih, wihBf, nIh);
        cvt_bf16_kernel<<<(int)(nHh  / 8 / 256), 256, 0, stream>>>(w_hh, whhBf, nHh);
        cvtT_bf16_kernel<<<dim3(2 * HH / 32, HH / 32), 256, 0, stream>>>(W1, W1T, 2 * HH, HH);
        cvtT_bf16_kernel<<<dim3(HH / 32, VV / 32), 256, 0, stream>>>(W2, W2T, HH, VV);
    }

    // gates_x = emb[tokens] @ w_ih^T + b_ih   (M=2048,K=512,N=3072)
    {
        dim3 g(TT * BB / BM, 3 * HH / BN);
        if (useBf)
            gemm_wmma_kernel<1,1,1,0><<<g,128,0,stream>>>(emb, EE, tokens, embBf,
                w_ih, EE, 1, wihBf, b_ih, gx, 3*HH, TT*BB, 3*HH, EE);
        else
            gemm_wmma_kernel<0,0,1,0><<<g,128,0,stream>>>(emb, EE, tokens, nullptr,
                w_ih, EE, 1, nullptr, b_ih, gx, 3*HH, TT*BB, 3*HH, EE);
    }

    for (int t = 0; t < TT; ++t) {
        {   // gh = h @ w_hh^T + b_hh  (M=64,K=1024,N=3072)
            dim3 g(1, 3 * HH / BN);
            if (useBf)
                gemm_wmma_kernel<0,1,0,0><<<g,128,0,stream>>>(h, HH, nullptr, nullptr,
                    w_hh, HH, 1, whhBf, b_hh, gh, 3*HH, BB, 3*HH, HH);
            else
                gemm_wmma_kernel<0,0,0,0><<<g,128,0,stream>>>(h, HH, nullptr, nullptr,
                    w_hh, HH, 1, nullptr, b_hh, gh, 3*HH, BB, 3*HH, HH);
        }
        gru_combine_kernel<<<BB * HH / 256, 256, 0, stream>>>(
            gx + (long)t * BB * 3 * HH, gh, h);
        attention_kernel<<<BB, 256, 0, stream>>>(h, enc, feat);
        bn_kernel<<<(2 * HH) / 256, 256, 0, stream>>>(feat, featn, bn1_g, bn1_b, 2 * HH);
        {   // y = ELU(featn @ W1 + b1)  (M=64,K=2048,N=1024)
            dim3 g(1, HH / BN);
            if (useBf)
                gemm_wmma_kernel<0,1,0,1><<<g,128,0,stream>>>(featn, 2*HH, nullptr, nullptr,
                    W1, HH, 0, W1T, b1, y, HH, BB, HH, 2*HH);
            else
                gemm_wmma_kernel<0,0,0,1><<<g,128,0,stream>>>(featn, 2*HH, nullptr, nullptr,
                    W1, HH, 0, nullptr, b1, y, HH, BB, HH, 2*HH);
        }
        bn_kernel<<<HH / 256, 256, 0, stream>>>(y, yn, bn2_g, bn2_b, HH);
        {   // logits = yn @ W2 + b2  (M=64,K=1024,N=32000)
            dim3 g(1, VV / BN);
            if (useBf)
                gemm_wmma_kernel<0,1,0,0><<<g,128,0,stream>>>(yn, HH, nullptr, nullptr,
                    W2, VV, 0, W2T, b2, logits, VV, BB, VV, HH);
            else
                gemm_wmma_kernel<0,0,0,0><<<g,128,0,stream>>>(yn, HH, nullptr, nullptr,
                    W2, VV, 0, nullptr, b2, logits, VV, BB, VV, HH);
        }
        softmax_rows_kernel<<<BB, 256, 0, stream>>>(logits, out, t);
    }

    hipMemcpyAsync(out + (size_t)BB * TT * VV, h,
                   (size_t)BB * HH * sizeof(float),
                   hipMemcpyDeviceToDevice, stream);
}
